// LDGCNN_semseg_s3dis_61890478735423
// MI455X (gfx1250) — matrix operand order repaired and checked by reading, not compiled, not run on hardware
//
#include <hip/hip_runtime.h>

typedef __bf16 bf16;
typedef float v2f   __attribute__((ext_vector_type(2)));
typedef float v8f   __attribute__((ext_vector_type(8)));
typedef bf16  v16bf __attribute__((ext_vector_type(16)));

#define B_  8
#define N_  4096
#define K_  20
#define CF_ 201   // channels in running feature tensor F

__device__ __forceinline__ float leaky(float x){ return x > 0.f ? x : 0.2f * x; }

// CAS-based float max; works on LDS and global generic pointers; order-independent result.
__device__ __forceinline__ void atomic_fmax(float* p, float v){
  int* ip = (int*)p;
  int old = __float_as_int(*p);
  while (__int_as_float(old) < v){
    int prev = atomicCAS(ip, old, __float_as_int(v));
    if (prev == old) break;
    old = prev;
  }
}

// 32-byte fragment load (handles 16B-aligned LDS addresses safely).
__device__ __forceinline__ v16bf load16bf(const bf16* p){
  v16bf r;
  __builtin_memcpy(&r, p, sizeof(r));
  return r;
}

__device__ __forceinline__ v2f load2f(const float* p){
  v2f r;
  __builtin_memcpy(&r, p, sizeof(r));
  return r;
}

// ---------------------------------------------------------------------------
// Pack f32 weights (scale-folded) into the WMMA A-fragment lane layout:
// linear index = ((mg*KC + kc)*32 + lane)*16 + e
// lane L holds row m = mg*16 + (L&15); element e -> v=e>>1, lo=e&1, hi=L>>4,
// k = kc*32 + hi*16 + e   (contiguous k order; B fragments in LDS use the same)
// ---------------------------------------------------------------------------
__global__ void pack_w(const float* __restrict__ w, const float* __restrict__ s,
                       bf16* __restrict__ out, int O, int C, int MT, int KC){
  int tot = MT * KC * 512;
  for (int idx = blockIdx.x * blockDim.x + threadIdx.x; idx < tot;
       idx += gridDim.x * blockDim.x){
    int e    = idx & 15;
    int lane = (idx >> 4) & 31;
    int rest = idx >> 9;
    int kc = rest % KC;
    int mg = rest / KC;
    int v = e >> 1, lo = e & 1, hi = lane >> 4;
    int k = kc * 32 + hi * 16 + e;
    int m = mg * 16 + (lane & 15);
    float val = 0.f;
    if (m < O && k < C){
      val = w[(size_t)m * C + k];
      if (s) val *= s[m];
    }
    out[idx] = (bf16)val;
  }
}

__global__ void copyx_kernel(const float* __restrict__ x, float* __restrict__ F){
  int t = blockIdx.x * blockDim.x + threadIdx.x;
  if (t >= B_ * 9 * N_) return;
  int b = t / (9 * N_), r = t % (9 * N_);
  F[(size_t)b * CF_ * N_ + r] = x[t];
}

// Build point-major, channel-padded KNN slice XT[b][n][Cp4] (zero padded) and
// squared norms in one pass.  Removes all bounds guards from the KNN hot loop.
__global__ void knnprep_kernel(const float* __restrict__ F, float* __restrict__ XT,
                               float* __restrict__ xx, int cs, int C, int Cp4){
  int t = blockIdx.x * blockDim.x + threadIdx.x;
  if (t >= B_ * N_) return;
  int b = t / N_, n = t % N_;
  float* dst = XT + (size_t)t * Cp4;
  float s = 0.f;
  for (int c = 0; c < Cp4; ++c){
    float v = (c < C) ? F[((size_t)b * CF_ + cs + c) * N_ + n] : 0.f;
    dst[c] = v;
    s += v * v;
  }
  xx[t] = s;
}

__global__ void ginit_kernel(float* __restrict__ g){
  int t = blockIdx.x * blockDim.x + threadIdx.x;
  if (t < B_ * 1024) g[t] = -3.0e38f;
}

// ---------------------------------------------------------------------------
// Fused KNN: one wave per 16-row tile. Inner products via V_WMMA_F32_16X16X4_F32
// (full f32 so neighbor ordering matches the f32 reference). Columns streamed
// from the padded, L2-resident XT buffer with unconditional b64 loads; running
// top-20 per row kept in LDS — the 537MB N x N distance matrix never exists.
// ---------------------------------------------------------------------------
__global__ void knn_kernel(const float* __restrict__ XT, const float* __restrict__ xx,
                           int* __restrict__ idxout, int Cp4){
  __shared__ __attribute__((aligned(32))) float sFeat[16 * 140];
  __shared__ float sXr[16];
  __shared__ float sDist[16 * 16];
  __shared__ float sTopV[16 * K_];
  __shared__ int   sTopI[16 * K_];
  const int lane = threadIdx.x;            // single wave32 block
  const int b  = blockIdx.y;
  const int n0 = blockIdx.x * 16;
  const int mcol = lane & 15;              // A-row / B-col slot for this lane
  const int hi   = lane >> 4;

  for (int t = lane; t < 16 * Cp4; t += 32){
    int m = t / Cp4, c = t % Cp4;
    sFeat[m * 140 + c] = XT[((size_t)b * N_ + n0 + m) * Cp4 + c];
  }
  if (lane < 16) sXr[lane] = xx[b * N_ + n0 + lane];
  for (int t = lane; t < 16 * K_; t += 32){ sTopV[t] = -3.0e38f; sTopI[t] = 0; }
  __syncthreads();

  float curMin = -3.0e38f;
  int   minSlot = 0;

  for (int nt = 0; nt < N_ / 16; ++nt){
    const int nc = nt * 16 + mcol;
    const float* bp = XT + ((size_t)b * N_ + nc) * Cp4 + 2 * hi;
    const float xxc = xx[b * N_ + nc];
    if (nt + 1 < N_ / 16)
      __builtin_prefetch(XT + ((size_t)b * N_ + nc + 16) * Cp4, 0, 0);
    v8f acc = {};
    for (int k0 = 0; k0 < Cp4; k0 += 4){
      v2f a  = load2f(&sFeat[mcol * 140 + k0 + 2 * hi]);
      v2f bb = load2f(bp + k0);
      acc = __builtin_amdgcn_wmma_f32_16x16x4_f32(false, a, false, bb,
                                                  (short)0, acc, false, false);
    }
#pragma unroll
    for (int v = 0; v < 8; ++v){
      int row = v + 8 * hi;
      sDist[row * 16 + mcol] = 2.f * acc[v] - sXr[row] - xxc;
    }
    __syncthreads();
    if (lane < 16){
#pragma unroll
      for (int j = 0; j < 16; ++j){
        float val = sDist[mcol * 16 + j];
        if (val > curMin){
          sTopV[mcol * K_ + minSlot] = val;
          sTopI[mcol * K_ + minSlot] = nt * 16 + j;
          float mn = sTopV[mcol * K_]; int ms = 0;
#pragma unroll
          for (int q = 1; q < K_; ++q){
            float tv = sTopV[mcol * K_ + q];
            if (tv < mn){ mn = tv; ms = q; }
          }
          curMin = mn; minSlot = ms;
        }
      }
    }
    __syncthreads();
  }
  if (lane < 16){
    for (int j = 0; j < K_; ++j)
      idxout[((size_t)b * N_ + n0 + mcol) * K_ + j] = sTopI[mcol * K_ + j];
  }
}

// ---------------------------------------------------------------------------
// Fused EdgeConv: 8 points/block. Gather neighbors -> edge features (bf16, LDS,
// 32-K chunks) -> conv1 (bf16 WMMA, scale folded) -> leaky -> LDS -> conv2
// (bf16 WMMA) -> leaky -> max over k via LDS fmax -> write x_i into F.
// Edge tensor (up to 718MB if materialized) never leaves LDS.
// ---------------------------------------------------------------------------
__global__ void edgeconv_kernel(const float* __restrict__ F, const int* __restrict__ idx,
                                const bf16* __restrict__ Wa, const float* __restrict__ ba,
                                const bf16* __restrict__ Wb, const float* __restrict__ bb,
                                float* __restrict__ Fout, int C, int KC1, int outOff){
  __shared__ int   sIdx[8 * K_];
  __shared__ float sCen[8 * 140];
  __shared__ __attribute__((aligned(32))) bf16 sE[160 * 40];  // [col][k], padded stride
  __shared__ __attribute__((aligned(32))) bf16 sH[160 * 72];  // [col][k], padded stride
  __shared__ float sMax[64 * 8];
  const int tid  = threadIdx.x;
  const int wave = tid >> 5, lane = tid & 31;
  const int half = lane >> 4, lcol = lane & 15;
  const int b  = blockIdx.y;
  const int n0 = blockIdx.x * 8;

  for (int t = tid; t < 8 * K_; t += 256)
    sIdx[t] = idx[((size_t)b * N_ + n0 + t / K_) * K_ + t % K_];
  for (int t = tid; t < 8 * C; t += 256){
    int p = t / C, c = t % C;
    sCen[p * 140 + c] = F[((size_t)b * CF_ + c) * N_ + n0 + p];
  }
  for (int t = tid; t < 512; t += 256) sMax[t] = -3.0e38f;
  __syncthreads();

  v8f acc[5] = {};
  for (int kc = 0; kc < KC1; ++kc){
    for (int e = tid; e < 160 * 32; e += 256){
      int cl = e >> 5, kk = e & 31;
      int k = kc * 32 + kk;
      int p = cl / K_;
      float val = 0.f;
      if (k < C)          val = F[((size_t)b * CF_ + k) * N_ + sIdx[cl]] - sCen[p * 140 + k];
      else if (k < 2 * C) val = sCen[p * 140 + (k - C)];
      sE[cl * 40 + kk] = (bf16)val;
    }
    __syncthreads();
#pragma unroll
    for (int i = 0; i < 5; ++i){
      int tileId = wave * 5 + i;
      int mt = tileId / 10, nt = tileId % 10;
      v16bf av = load16bf(Wa + ((size_t)(mt * KC1 + kc) * 32 + lane) * 16);
      v16bf bv = load16bf(sE + (nt * 16 + lcol) * 40 + half * 16);
      acc[i] = __builtin_amdgcn_wmma_f32_16x16x32_bf16(false, av, false, bv,
                                                       (short)0, acc[i], false, false);
    }
    __syncthreads();
  }
#pragma unroll
  for (int i = 0; i < 5; ++i){
    int tileId = wave * 5 + i;
    int mt = tileId / 10, nt = tileId % 10;
    int cl = nt * 16 + lcol;
#pragma unroll
    for (int v = 0; v < 8; ++v){
      int row = mt * 16 + v + 8 * half;
      sH[cl * 72 + row] = (bf16)leaky(acc[i][v] + ba[row]);
    }
  }
  __syncthreads();

  v8f acc2[5] = {};
#pragma unroll
  for (int kc = 0; kc < 2; ++kc){
#pragma unroll
    for (int i = 0; i < 5; ++i){
      int tileId = wave * 5 + i;
      int mt = tileId / 10, nt = tileId % 10;
      v16bf av = load16bf(Wb + ((size_t)(mt * 2 + kc) * 32 + lane) * 16);
      v16bf bv = load16bf(sH + (nt * 16 + lcol) * 72 + kc * 32 + half * 16);
      acc2[i] = __builtin_amdgcn_wmma_f32_16x16x32_bf16(false, av, false, bv,
                                                        (short)0, acc2[i], false, false);
    }
  }
#pragma unroll
  for (int i = 0; i < 5; ++i){
    int tileId = wave * 5 + i;
    int mt = tileId / 10, nt = tileId % 10;
    int cl = nt * 16 + lcol;
    int p  = cl / K_;
#pragma unroll
    for (int v = 0; v < 8; ++v){
      int row = mt * 16 + v + 8 * half;
      atomic_fmax(&sMax[row * 8 + p], leaky(acc2[i][v] + bb[row]));
    }
  }
  __syncthreads();
  for (int t = tid; t < 512; t += 256){
    int ch = t >> 3, p = t & 7;
    Fout[((size_t)b * CF_ + outOff + ch) * N_ + n0 + p] = sMax[ch * 8 + p];
  }
}

// ---------------------------------------------------------------------------
// Generic 1x1-conv GEMM: Y = epilogue((s-folded W) X + b). 64x128 block tile,
// 8 waves x 4 accum tiles, K in 32-chunks staged through padded LDS as bf16.
// xmode: 0 = F(f32, 201ch), 1 = [g(1024) | F ch9..200], 2 = bf16 buffer
// omode: 0 = leaky+max-over-n atomically into g, 1 = leaky->bf16 buffer,
//        2 = raw f32 output (final w12)
// ---------------------------------------------------------------------------
__global__ void gemm_kernel(const bf16* __restrict__ Wp, const float* __restrict__ bias,
                            const float* __restrict__ F, const float* __restrict__ g_in,
                            const bf16* __restrict__ Xb,
                            float* __restrict__ g_out, bf16* __restrict__ Yout,
                            float* __restrict__ Oout,
                            int O, int C, int KC, int mtiles, int xmode, int omode){
  __shared__ __attribute__((aligned(32))) bf16 sX[128 * 40];
  const int tid = threadIdx.x, wave = tid >> 5, lane = tid & 31;
  const int half = lane >> 4, lcol = lane & 15;
  const int b = blockIdx.z, by = blockIdx.y, bx = blockIdx.x;

  v8f acc[4] = {};
  for (int kc = 0; kc < KC; ++kc){
    for (int e = tid; e < 128 * 32; e += 256){
      int cl = e >> 5, kk = e & 31;
      int k = kc * 32 + kk;
      int n = bx * 128 + cl;
      float val = 0.f;
      if (xmode == 0){
        if (k < C) val = F[((size_t)b * CF_ + k) * N_ + n];
      } else if (xmode == 1){
        if (k < 1024)    val = g_in[b * 1024 + k];
        else if (k < C)  val = F[((size_t)b * CF_ + 9 + (k - 1024)) * N_ + n];
      } else {
        if (k < C) val = (float)Xb[((size_t)b * C + k) * N_ + n];
      }
      sX[cl * 40 + kk] = (bf16)val;
    }
    __syncthreads();
    v16bf bv = load16bf(sX + (wave * 16 + lcol) * 40 + half * 16);
#pragma unroll
    for (int i = 0; i < 4; ++i){
      if (i < mtiles){
        int mg = by * 4 + i;
        v16bf av = load16bf(Wp + ((size_t)(mg * KC + kc) * 32 + lane) * 16);
        acc[i] = __builtin_amdgcn_wmma_f32_16x16x32_bf16(false, av, false, bv,
                                                         (short)0, acc[i], false, false);
      }
    }
    __syncthreads();
  }

  const int n = bx * 128 + wave * 16 + lcol;
#pragma unroll
  for (int i = 0; i < 4; ++i){
    if (i < mtiles){
#pragma unroll
      for (int v = 0; v < 8; ++v){
        int o = by * 64 + i * 16 + v + 8 * half;
        float y = acc[i][v];
        if (omode == 0){
          y = leaky(y + bias[o]);
#pragma unroll
          for (int m = 1; m < 16; m <<= 1) y = fmaxf(y, __shfl_xor(y, m, 32));
          if (lcol == 0) atomic_fmax(&g_out[b * 1024 + o], y);
        } else if (omode == 1){
          y = leaky(y + bias[o]);
          Yout[((size_t)b * O + o) * N_ + n] = (bf16)y;
        } else {
          if (o < O) Oout[((size_t)b * O + o) * N_ + n] = y;
        }
      }
    }
  }
}

// ---------------------------------------------------------------------------
extern "C" void kernel_launch(void* const* d_in, const int* in_sizes, int n_in,
                              void* d_out, int out_size, void* d_ws, size_t ws_size,
                              hipStream_t stream){
  (void)in_sizes; (void)n_in; (void)out_size; (void)ws_size;
  const float* x   = (const float*)d_in[0];
  const float* w1  = (const float*)d_in[1];  const float* s1  = (const float*)d_in[2];  const float* b1  = (const float*)d_in[3];
  const float* w2  = (const float*)d_in[4];  const float* s2  = (const float*)d_in[5];  const float* b2  = (const float*)d_in[6];
  const float* w3  = (const float*)d_in[7];  const float* s3  = (const float*)d_in[8];  const float* b3  = (const float*)d_in[9];
  const float* w4  = (const float*)d_in[10]; const float* s4  = (const float*)d_in[11]; const float* b4  = (const float*)d_in[12];
  const float* w5  = (const float*)d_in[13]; const float* s5  = (const float*)d_in[14]; const float* b5  = (const float*)d_in[15];
  const float* w6  = (const float*)d_in[16]; const float* s6  = (const float*)d_in[17]; const float* b6  = (const float*)d_in[18];
  const float* w9  = (const float*)d_in[19]; const float* s9  = (const float*)d_in[20]; const float* b9  = (const float*)d_in[21];
  const float* w10 = (const float*)d_in[22]; const float* s10 = (const float*)d_in[23]; const float* b10 = (const float*)d_in[24];
  const float* w11 = (const float*)d_in[25]; const float* s11 = (const float*)d_in[26]; const float* b11 = (const float*)d_in[27];
  const float* w12 = (const float*)d_in[28];

  char* ws = (char*)d_ws;
  size_t off = 0;
  auto take = [&](size_t bytes) -> char* {
    char* p = ws + off;
    off = (off + bytes + 255) & ~(size_t)255;
    return p;
  };
  float* F   = (float*)take(sizeof(float) * (size_t)B_ * CF_ * N_);
  float* XT  = (float*)take(sizeof(float) * (size_t)B_ * N_ * 140);
  float* XX  = (float*)take(sizeof(float) * (size_t)B_ * N_);
  int*   IDX = (int*)  take(sizeof(int)   * (size_t)B_ * N_ * K_);
  auto pwsz = [](int O, int C) -> size_t {
    return (size_t)((O + 15) / 16) * ((C + 31) / 32) * 512;
  };
  bf16* pw1  = (bf16*)take(2 * pwsz(64, 18));
  bf16* pw2  = (bf16*)take(2 * pwsz(64, 64));
  bf16* pw3  = (bf16*)take(2 * pwsz(64, 146));
  bf16* pw4  = (bf16*)take(2 * pwsz(64, 64));
  bf16* pw5  = (bf16*)take(2 * pwsz(64, 274));
  bf16* pw6  = (bf16*)take(2 * pwsz(64, 64));
  bf16* pw9  = (bf16*)take(2 * pwsz(1024, 201));
  bf16* pw10 = (bf16*)take(2 * pwsz(512, 1216));
  bf16* pw11 = (bf16*)take(2 * pwsz(256, 512));
  bf16* pw12 = (bf16*)take(2 * pwsz(16, 256));
  float* G   = (float*)take(sizeof(float) * (size_t)B_ * 1024);
  bf16* Y10  = (bf16*)take(2 * (size_t)B_ * 512 * N_);
  bf16* Y11  = (bf16*)take(2 * (size_t)B_ * 256 * N_);

  auto packLaunch = [&](const float* w, const float* s, bf16* outp, int O, int C){
    int MT = (O + 15) / 16, KC = (C + 31) / 32;
    int tot = MT * KC * 512;
    pack_w<<<dim3((tot + 255) / 256), dim3(256), 0, stream>>>(w, s, outp, O, C, MT, KC);
  };
  packLaunch(w1,  s1,  pw1,  64,   18);
  packLaunch(w2,  s2,  pw2,  64,   64);
  packLaunch(w3,  s3,  pw3,  64,   146);
  packLaunch(w4,  s4,  pw4,  64,   64);
  packLaunch(w5,  s5,  pw5,  64,   274);
  packLaunch(w6,  s6,  pw6,  64,   64);
  packLaunch(w9,  s9,  pw9,  1024, 201);
  packLaunch(w10, s10, pw10, 512,  1216);
  packLaunch(w11, s11, pw11, 256,  512);
  packLaunch(w12, nullptr, pw12, 13, 256);

  copyx_kernel<<<dim3((B_ * 9 * N_ + 255) / 256), dim3(256), 0, stream>>>(x, F);

  const int nBN = (B_ * N_ + 255) / 256;

  // Stage 1: knn over channels 6..8, edgeconv C=9 -> x1 at F[9..72]
  knnprep_kernel<<<dim3(nBN), dim3(256), 0, stream>>>(F, XT, XX, 6, 3, 4);
  knn_kernel<<<dim3(N_ / 16, B_), dim3(32), 0, stream>>>(XT, XX, IDX, 4);
  edgeconv_kernel<<<dim3(N_ / 8, B_), dim3(256), 0, stream>>>(F, IDX, pw1, b1, pw2, b2, F, 9, 1, 9);

  // Stage 2: knn over channels 0..72, edgeconv C=73 -> x2 at F[73..136]
  knnprep_kernel<<<dim3(nBN), dim3(256), 0, stream>>>(F, XT, XX, 0, 73, 76);
  knn_kernel<<<dim3(N_ / 16, B_), dim3(32), 0, stream>>>(XT, XX, IDX, 76);
  edgeconv_kernel<<<dim3(N_ / 8, B_), dim3(256), 0, stream>>>(F, IDX, pw3, b3, pw4, b4, F, 73, 5, 73);

  // Stage 3: knn over channels 0..136, edgeconv C=137 -> x3 at F[137..200]
  knnprep_kernel<<<dim3(nBN), dim3(256), 0, stream>>>(F, XT, XX, 0, 137, 140);
  knn_kernel<<<dim3(N_ / 16, B_), dim3(32), 0, stream>>>(XT, XX, IDX, 140);
  edgeconv_kernel<<<dim3(N_ / 8, B_), dim3(256), 0, stream>>>(F, IDX, pw5, b5, pw6, b6, F, 137, 9, 137);

  // w9: global feature g = max_n leaky(W9 f + b9)  (fused max, y9 never stored)
  ginit_kernel<<<dim3((B_ * 1024 + 255) / 256), dim3(256), 0, stream>>>(G);
  gemm_kernel<<<dim3(N_ / 128, 1024 / 64, B_), dim3(256), 0, stream>>>(
      pw9, b9, F, nullptr, nullptr, G, nullptr, nullptr, 1024, 201, 7, 4, 0, 0);

  // w10: [g | x1 x2 x3] (1216ch) -> Y10 (bf16)
  gemm_kernel<<<dim3(N_ / 128, 512 / 64, B_), dim3(256), 0, stream>>>(
      pw10, b10, F, G, nullptr, nullptr, Y10, nullptr, 512, 1216, 38, 4, 1, 1);

  // w11: Y10 -> Y11 (bf16)
  gemm_kernel<<<dim3(N_ / 128, 256 / 64, B_), dim3(256), 0, stream>>>(
      pw11, b11, nullptr, nullptr, Y10, nullptr, Y11, nullptr, 256, 512, 16, 4, 2, 1);

  // w12: Y11 -> output (f32, no activation)
  gemm_kernel<<<dim3(N_ / 128, 1, B_), dim3(256), 0, stream>>>(
      pw12, nullptr, nullptr, nullptr, Y11, nullptr, nullptr, (float*)d_out,
      13, 256, 8, 1, 2, 2);
}
